// MultiHeadAttention_11381663334841
// MI455X (gfx1250) — matrix operand order repaired and checked by reading, not compiled, not run on hardware
//
#include <hip/hip_runtime.h>
#include <hip/hip_bf16.h>

typedef _Float16 h8  __attribute__((ext_vector_type(8)));
typedef _Float16 h16 __attribute__((ext_vector_type(16)));
typedef float    f8  __attribute__((ext_vector_type(8)));

// Problem constants (B,T,D,H) = (2,2048,1024,16), DH=64
#define TB 2
#define TT 2048
#define TD 1024
#define TH 16
#define TDH 64

__device__ __forceinline__ f8 wmma_f16(h16 a, h16 b, f8 c) {
  // D = A(16x32 f16) * B(32x16 f16) + C(16x16 f32)
  return __builtin_amdgcn_wmma_f32_16x16x32_f16(false, a, false, b, (short)0, c,
                                                false, false);
}

__device__ __forceinline__ h16 cat8(h8 lo, h8 hi) {
  return __builtin_shufflevector(lo, hi, 0, 1, 2, 3, 4, 5, 6, 7, 8, 9, 10, 11,
                                 12, 13, 14, 15);
}

// A fragment 16x32 f16 from row-major matrix (leading dim ld, elements).
// ISA layout: lanes 0-15 -> row M=lane, K chunks [0..7] and [16..23];
//             lanes 16-31 -> row M=lane-16, K chunks [8..15] and [24..31].
__device__ __forceinline__ h16 load_a16x32(const _Float16* p, int ld, int row0,
                                           int k0) {
  int lane = threadIdx.x & 31;
  const _Float16* q =
      p + (size_t)(row0 + (lane & 15)) * ld + k0 + ((lane >> 4) << 3);
  h8 lo = *(const h8*)q;
  h8 hi = *(const h8*)(q + 16);
  return cat8(lo, hi);
}

// B fragment 32x16 f16; column n of B is contiguous in k at p + n*ld + k.
// ISA layout: lanes 0-15 -> col N=lane, K [0..7],[16..23]; lanes 16-31 -> K [8..15],[24..31].
__device__ __forceinline__ h16 load_b32x16(const _Float16* p, int ld, int n0,
                                           int k0) {
  int lane = threadIdx.x & 31;
  const _Float16* q =
      p + (size_t)(n0 + (lane & 15)) * ld + k0 + ((lane >> 4) << 3);
  h8 lo = *(const h8*)q;
  h8 hi = *(const h8*)(q + 16);
  return cat8(lo, hi);
}

__global__ void __launch_bounds__(256) cvt_f32_f16(const float* __restrict__ in,
                                                   _Float16* __restrict__ out,
                                                   int n) {
  int i = blockIdx.x * blockDim.x + threadIdx.x;
  if (i < n) out[i] = (_Float16)in[i];
}

// C[M,N] (f32, row-major) = A[M,K] (f16 row-major) * W[N,K]^T (f16 row-major)
// Per wave: 16x32 output tile. Block = 8 waves stacked in M (128 rows).
__global__ void __launch_bounds__(256) gemm_wmma(const _Float16* __restrict__ A,
                                                 const _Float16* __restrict__ W,
                                                 float* __restrict__ C, int N,
                                                 int K) {
  int wave = threadIdx.x >> 5;
  int lane = threadIdx.x & 31;
  int m0 = blockIdx.x * 128 + wave * 16;
  int n0 = blockIdx.y * 32;
  f8 acc0 = {}, acc1 = {};
  for (int k = 0; k < K; k += 32) {
    h16 a = load_a16x32(A, K, m0, k);
    h16 b0 = load_b32x16(W, K, n0, k);
    h16 b1 = load_b32x16(W, K, n0 + 16, k);
    acc0 = wmma_f16(a, b0, acc0);
    acc1 = wmma_f16(a, b1, acc1);
  }
  // C layout: lanes 0-15 hold (M=v, N=lane); lanes 16-31 hold (M=8+v, N=lane-16)
  int hf = (lane >> 4) << 3;
  int nn = lane & 15;
#pragma unroll
  for (int v = 0; v < 8; ++v) {
    size_t r = (size_t)(m0 + hf + v) * N;
    C[r + n0 + nn] = acc0[v];
    C[r + n0 + 16 + nn] = acc1[v];
  }
}

// RoPE + head split. qkv is [B*T, 3*D] f32.
// Emits Qh,Kh as [b,h,t,dh] f16 (RoPE applied) and Vt as [b,h,dh,t] f16.
__global__ void __launch_bounds__(256) rope_split(
    const float* __restrict__ qkv, const int* __restrict__ pos,
    const float* __restrict__ cosc, const float* __restrict__ sinc,
    _Float16* __restrict__ Qh, _Float16* __restrict__ Kh,
    _Float16* __restrict__ Vt) {
  int idx = blockIdx.x * blockDim.x + threadIdx.x;  // B*H*T*(DH/2) = 2^21
  int i = idx & 31;            // rotary pair index
  int t = (idx >> 5) & 2047;   // token
  int h = (idx >> 16) & 15;    // head
  int b = idx >> 20;           // batch
  int p = pos[t];
  float c = cosc[p * 32 + i];
  float s = sinc[p * 32 + i];
  const float* qp = qkv + (size_t)(b * TT + t) * (3 * TD) + h * TDH + 2 * i;
  float q1 = qp[0], q2 = qp[1];
  float k1 = qp[TD], k2 = qp[TD + 1];
  float v1 = qp[2 * TD], v2 = qp[2 * TD + 1];
  size_t hb = ((size_t)(b * TH + h) * TT + t) * TDH + 2 * i;
  Qh[hb] = (_Float16)(q1 * c - q2 * s);
  Qh[hb + 1] = (_Float16)(q1 * s + q2 * c);
  Kh[hb] = (_Float16)(k1 * c - k2 * s);
  Kh[hb + 1] = (_Float16)(k1 * s + k2 * c);
  size_t vb = ((size_t)(b * TH + h) * TDH + 2 * i) * TT + t;
  Vt[vb] = (_Float16)v1;
  Vt[vb + TT] = (_Float16)v2;
}

// Flash attention: one 16-row q-tile per wave, 32-key chunks, causal.
__global__ void __launch_bounds__(256) attn_wmma(const _Float16* __restrict__ Qh,
                                                 const _Float16* __restrict__ Kh,
                                                 const _Float16* __restrict__ Vt,
                                                 _Float16* __restrict__ Oa) {
  // per-wave P staging: 16 rows x 32 f16, padded row stride 40 (80B) to keep
  // ds_load_b128 16B-aligned and bank-conflict free.
  __shared__ _Float16 pbuf[8][16 * 40];
  int wave = threadIdx.x >> 5;
  int lane = threadIdx.x & 31;
  int w = blockIdx.x * 8 + wave;
  int qt = w & 127;         // T/16 = 128 q-tiles
  int h = (w >> 7) & 15;    // head
  int b = w >> 11;          // batch
  int q0 = qt * 16;
  const _Float16* Qp = Qh + (size_t)(b * TH + h) * TT * TDH;
  const _Float16* Kp = Kh + (size_t)(b * TH + h) * TT * TDH;
  const _Float16* Vp = Vt + (size_t)(b * TH + h) * TDH * TT;
  _Float16* pw = &pbuf[wave][0];

  // Q fragments (constant across k loop): 16x64 -> two 16x32 A frags
  h16 qa0 = load_a16x32(Qp, TDH, q0, 0);
  h16 qa1 = load_a16x32(Qp, TDH, q0, 32);

  f8 o0 = {}, o1 = {}, o2 = {}, o3 = {};
  float m[8], lsum[8];
#pragma unroll
  for (int v = 0; v < 8; ++v) {
    m[v] = -1e30f;
    lsum[v] = 0.0f;
  }

  int hf = (lane >> 4) << 3;  // row offset of this half-wave in C layout
  int nn = lane & 15;         // column within tile
  int kend = q0 + 16;         // keys needed: [0, q0+16)
  for (int k0 = 0; k0 < kend; k0 += 32) {
    // S = Q * K^T for keys [k0, k0+32): two 16x16 C tiles
    h16 kb00 = load_b32x16(Kp, TDH, k0, 0);
    h16 kb01 = load_b32x16(Kp, TDH, k0, 32);
    h16 kb10 = load_b32x16(Kp, TDH, k0 + 16, 0);
    h16 kb11 = load_b32x16(Kp, TDH, k0 + 16, 32);
    f8 s0 = {}, s1 = {};
    s0 = wmma_f16(qa0, kb00, s0);
    s0 = wmma_f16(qa1, kb01, s0);
    s1 = wmma_f16(qa0, kb10, s1);
    s1 = wmma_f16(qa1, kb11, s1);

    float corr[8];
#pragma unroll
    for (int v = 0; v < 8; ++v) {
      int qrow = q0 + hf + v;
      // branch-free causal mask (EXEC must stay all-ones around WMMA)
      float x0 = (k0 + nn <= qrow) ? s0[v] * 0.125f : -1e30f;
      float x1 = (k0 + 16 + nn <= qrow) ? s1[v] * 0.125f : -1e30f;
      // row-max across 16 lanes of this half (xor 1/2/4/8 never crosses halves)
      float cm = fmaxf(x0, x1);
      cm = fmaxf(cm, __shfl_xor(cm, 1, 32));
      cm = fmaxf(cm, __shfl_xor(cm, 2, 32));
      cm = fmaxf(cm, __shfl_xor(cm, 4, 32));
      cm = fmaxf(cm, __shfl_xor(cm, 8, 32));
      float nm = fmaxf(m[v], cm);
      float cr = __expf(m[v] - nm);
      m[v] = nm;
      float e0 = __expf(x0 - nm);
      float e1 = __expf(x1 - nm);
      float cs = e0 + e1;
      cs += __shfl_xor(cs, 1, 32);
      cs += __shfl_xor(cs, 2, 32);
      cs += __shfl_xor(cs, 4, 32);
      cs += __shfl_xor(cs, 8, 32);
      lsum[v] = lsum[v] * cr + cs;
      corr[v] = cr;
      // stage P (C layout -> row-major f16 in LDS)
      pw[(hf + v) * 40 + nn] = (_Float16)e0;
      pw[(hf + v) * 40 + 16 + nn] = (_Float16)e1;
    }
#pragma unroll
    for (int v = 0; v < 8; ++v) {
      o0[v] *= corr[v];
      o1[v] *= corr[v];
      o2[v] *= corr[v];
      o3[v] *= corr[v];
    }
    // DS ops are in-order within a wave; only compiler reordering must be
    // blocked (wave-divergent loop bound forbids __syncthreads here).
    asm volatile("" ::: "memory");
    {
      // reload P as 16x32 A fragment from LDS
      const _Float16* q = pw + (lane & 15) * 40 + ((lane >> 4) << 3);
      h8 lo = *(const h8*)q;
      h8 hi = *(const h8*)(q + 16);
      h16 pa = cat8(lo, hi);
      // V fragments: Vt is [dh, t], so B columns (n=dh) are contiguous in keys
      h16 vb0 = load_b32x16(Vp, TT, 0, k0);
      h16 vb1 = load_b32x16(Vp, TT, 16, k0);
      h16 vb2 = load_b32x16(Vp, TT, 32, k0);
      h16 vb3 = load_b32x16(Vp, TT, 48, k0);
      o0 = wmma_f16(pa, vb0, o0);
      o1 = wmma_f16(pa, vb1, o1);
      o2 = wmma_f16(pa, vb2, o2);
      o3 = wmma_f16(pa, vb3, o3);
    }
    asm volatile("" ::: "memory");
  }
  // normalize and store to [B*T, D] f16 (row-major) for the output projection
#pragma unroll
  for (int v = 0; v < 8; ++v) {
    float inv = 1.0f / lsum[v];
    size_t row = (size_t)(b * TT + q0 + hf + v) * TD + h * TDH + nn;
    Oa[row] = (_Float16)(o0[v] * inv);
    Oa[row + 16] = (_Float16)(o1[v] * inv);
    Oa[row + 32] = (_Float16)(o2[v] * inv);
    Oa[row + 48] = (_Float16)(o3[v] * inv);
  }
}

extern "C" void kernel_launch(void* const* d_in, const int* in_sizes, int n_in,
                              void* d_out, int out_size, void* d_ws,
                              size_t ws_size, hipStream_t stream) {
  (void)in_sizes;
  (void)n_in;
  (void)out_size;
  (void)ws_size;
  const float* x = (const float*)d_in[0];
  const int* tpos = (const int*)d_in[1];
  const float* Wqkv = (const float*)d_in[2];
  const float* Wo = (const float*)d_in[3];
  const float* cosc = (const float*)d_in[4];
  const float* sinc = (const float*)d_in[5];
  float* out = (float*)d_out;

  char* ws = (char*)d_ws;
  _Float16* xh = (_Float16*)(ws);                      //  8 MB  x f16
  _Float16* wqh = (_Float16*)(ws + (8ull << 20));      //  6 MB  W_qkv f16
  _Float16* woh = (_Float16*)(ws + (14ull << 20));     //  2 MB  W_o f16
  float* qkv = (float*)(ws + (16ull << 20));           // 48 MB  qkv f32
  _Float16* Qh = (_Float16*)(ws + (64ull << 20));      //  8 MB
  _Float16* Kh = (_Float16*)(ws + (72ull << 20));      //  8 MB
  _Float16* Vt = (_Float16*)(ws + (80ull << 20));      //  8 MB
  _Float16* attn = (_Float16*)(ws + (88ull << 20));    //  8 MB (total 96 MB)

  const int NX = TB * TT * TD;        // 4,194,304
  const int NWQ = 3 * TD * TD;        // 3,145,728
  const int NWO = TD * TD;            // 1,048,576
  cvt_f32_f16<<<NX / 256, 256, 0, stream>>>(x, xh, NX);
  cvt_f32_f16<<<NWQ / 256, 256, 0, stream>>>(Wqkv, wqh, NWQ);
  cvt_f32_f16<<<NWO / 256, 256, 0, stream>>>(Wo, woh, NWO);

  // qkv[4096, 3072] = x[4096,1024] * W_qkv^T ; grid M/128 x N/32
  gemm_wmma<<<dim3(32, 96), 256, 0, stream>>>(xh, wqh, qkv, 3 * TD, TD);

  // RoPE + split: B*H*T*(DH/2) = 2^21 threads
  rope_split<<<(1u << 21) / 256, 256, 0, stream>>>(qkv, tpos, cosc, sinc, Qh,
                                                   Kh, Vt);

  // attention: B*H*(T/16) = 4096 waves, 8 waves/block
  attn_wmma<<<512, 256, 0, stream>>>(Qh, Kh, Vt, attn);

  // out[4096,1024] = attn[4096,1024] * W_o^T
  gemm_wmma<<<dim3(32, 32), 256, 0, stream>>>(attn, woh, out, TD, TD);
}
